// CrystalGraphConvNet_31980326486278
// MI455X (gfx1250) — matrix operand order repaired and checked by reading, not compiled, not run on hardware
//
#include <hip/hip_runtime.h>

typedef __attribute__((ext_vector_type(16))) _Float16 v16h;
typedef __attribute__((ext_vector_type(8)))  _Float16 v8h;
typedef __attribute__((ext_vector_type(8)))  float    v8f;

#define DINL __device__ __forceinline__

// ---------------- WMMA + helpers ----------------

DINL v8f wmma_f16(v16h a, v16h b, v8f c) {
  // v_wmma_f32_16x16x32_f16 : D = A(16x32 f16) * B(32x16 f16) + C(16x16 f32)
  return __builtin_amdgcn_wmma_f32_16x16x32_f16(false, a, false, b, (short)0, c,
                                                false, false);
}

DINL void lds_fence() {
  // CDNA5 split counters: wait for all outstanding DS ops of this wave so
  // cross-lane LDS staging is visible before WMMA fragment loads.
  asm volatile("s_wait_dscnt 0x0" ::: "memory");
}

DINL float elu_f(float x)      { return x > 0.f ? x : expm1f(x); }
DINL float sigmoid_f(float x)  { return 1.f / (1.f + expf(-x)); }
DINL float softplus_f(float x) { return fmaxf(x, 0.f) + log1pf(expf(-fabsf(x))); }

DINL v16h cat16(v8h lo, v8h hi8) {
  v16h r;
#pragma unroll
  for (int j = 0; j < 8; ++j) { r[j] = lo[j]; r[j + 8] = hi8[j]; }
  return r;
}

DINL v8f zero8f() {
  v8f z;
#pragma unroll
  for (int j = 0; j < 8; ++j) z[j] = 0.f;
  return z;
}

// ---------------- LDS tile staging ----------------
// Tile layout: padded image, position-major / channel-minor:
//   tile[(py*10+px)*16 + ci], py,px in [0,10), halo = zeros, f16.
// Interior pixel (y,x) lives at py=y+1, px=x+1.

DINL void zero_tile(_Float16* tile, int lane) {
  v8h z;
#pragma unroll
  for (int j = 0; j < 8; ++j) z[j] = (_Float16)0.f;
  for (int c = lane; c < 200; c += 32)           // 200 * 8 halfs = 1600
    *(v8h*)(tile + c * 8) = z;
}

DINL void stage_tile(_Float16* tile, const float* __restrict__ img, int lane) {
  zero_tile(tile, lane);
  const float4* img4 = (const float4*)img;       // (16,8,8) f32 = 256 float4
#pragma unroll
  for (int i = 0; i < 8; ++i) {
    float4 v = img4[lane + 32 * i];
    int e   = (lane + 32 * i) * 4;               // element index, 4 consecutive x
    int ci  = e >> 6;
    int pos = e & 63;
    int y = pos >> 3, x = pos & 7;
    _Float16* p = tile + ((y + 1) * 10 + (x + 1)) * 16 + ci;
    p[0]  = (_Float16)v.x;                       // x+1 step = +16 halfs
    p[16] = (_Float16)v.y;
    p[32] = (_Float16)v.z;
    p[48] = (_Float16)v.w;
  }
}

// ---------------- WMMA fragment loads ----------------
// A (16x32 f16): lane<16 -> row M=lane, halfs 0..7 = K 0..7, 8..15 = K 16..23
//                lane>=16 -> row M=lane-16, K 8..15 and 24..31.
// With K = tap*16 + ci this is: chunk0 = (tap 2s, ci base), chunk1 = (tap 2s+1).
DINL v16h load_a_frag(const _Float16* tile, int lane, int mt, int s) {
  int pos = mt * 16 + (lane & 15);
  int cib = (lane >> 4) << 3;                    // ci base: 0 or 8
  int y = pos >> 3, x = pos & 7;
  int t0 = 2 * s;
  int t1 = t0 + 1;
  if (t1 > 8) t1 = 8;                            // tap 9: B frag is zero there
  const v8h* p0 = (const v8h*)(tile + ((y + t0 / 3) * 10 + (x + t0 % 3)) * 16 + cib);
  const v8h* p1 = (const v8h*)(tile + ((y + t1 / 3) * 10 + (x + t1 % 3)) * 16 + cib);
  return cat16(*p0, *p1);
}

// B fragments precomputed in workspace: fragbuf[(fid*32+lane)*16 + j]
DINL v16h load_bfrag(const _Float16* __restrict__ fragbuf, int fid, int lane) {
  const v8h* p = (const v8h*)(fragbuf + (size_t)(fid * 32 + lane) * 16);
  return cat16(p[0], p[1]);
}

// frag ids: 0..4 node_w, 5..9 edge_w, 10..14 lin_w (co 0..15), 15..19 lin_w (co 16..31)
DINL void fill_frag(_Float16* __restrict__ fragbuf, int fid,
                    const float* __restrict__ w, int nt, int s, int lane) {
  int co  = nt * 16 + (lane & 15);
  int cib = (lane >> 4) << 3;                    // ci base: 0 or 8
#pragma unroll
  for (int j = 0; j < 16; ++j) {                 // j<8: tap 2s, ci cib+j; else tap 2s+1, ci cib+j-8
    int tap = 2 * s + (j >> 3);
    float v = (tap < 9) ? w[(co * 16 + cib + (j & 7)) * 9 + tap] : 0.f;
    fragbuf[(size_t)(fid * 32 + lane) * 16 + j] = (_Float16)v;
  }
}

__global__ void frag_setup_kernel(const float* __restrict__ edge_w,
                                  const float* __restrict__ node_w,
                                  const float* __restrict__ lin_w,
                                  _Float16* __restrict__ fragbuf) {
  int lane = threadIdx.x;
  if (lane >= 32) return;
  for (int s = 0; s < 5; ++s) fill_frag(fragbuf,      s, node_w, 0, s, lane);
  for (int s = 0; s < 5; ++s) fill_frag(fragbuf,  5 + s, edge_w, 0, s, lane);
  for (int s = 0; s < 5; ++s) fill_frag(fragbuf, 10 + s, lin_w,  0, s, lane);
  for (int s = 0; s < 5; ++s) fill_frag(fragbuf, 15 + s, lin_w,  1, s, lane);
}

// ---------------- node conv: node_out = conv3x3(atom, node_w) ----------------
__global__ void __launch_bounds__(256)
node_conv_kernel(const float* __restrict__ atom,
                 const _Float16* __restrict__ fragbuf,
                 float* __restrict__ node_out, int N) {
  __shared__ alignas(16) _Float16 lds[8 * 1600];
  int wave = threadIdx.x >> 5, lane = threadIdx.x & 31;
  int n = blockIdx.x * 8 + wave;
  if (n >= N) return;
  _Float16* tile = lds + wave * 1600;
  stage_tile(tile, atom + (size_t)n * 1024, lane);
  v16h bf[5];
#pragma unroll
  for (int s = 0; s < 5; ++s) bf[s] = load_bfrag(fragbuf, s, lane);
  lds_fence();
  int co = lane & 15, hi = lane >> 4;
#pragma unroll
  for (int mt = 0; mt < 4; ++mt) {
    v8f acc = zero8f();
#pragma unroll
    for (int s = 0; s < 5; ++s)
      acc = wmma_f16(load_a_frag(tile, lane, mt, s), bf[s], acc);
    // D: lane holds channel co, 8 consecutive positions starting pos0
    int pos0 = mt * 16 + hi * 8;
    float* dst = node_out + ((size_t)n * 16 + co) * 64 + pos0;
    *(float4*)dst       = make_float4(acc[0], acc[1], acc[2], acc[3]);
    *(float4*)(dst + 4) = make_float4(acc[4], acc[5], acc[6], acc[7]);
  }
}

// ---------------- per-edge pipeline ----------------
__global__ void __launch_bounds__(256)
edge_kernel(const float* __restrict__ atom, const int* __restrict__ esrc,
            const int* __restrict__ etgt, const float* __restrict__ node_out,
            const float* __restrict__ lin_b, const _Float16* __restrict__ fragbuf,
            float* __restrict__ accum, int E) {
  __shared__ alignas(16) _Float16 lds[8 * 3200];
  int wave = threadIdx.x >> 5, lane = threadIdx.x & 31;
  int e = blockIdx.x * 8 + wave;
  if (e >= E) return;
  _Float16* tileA = lds + wave * 3200;
  _Float16* tileZ = tileA + 1600;
  int src = esrc[e], tgt = etgt[e];
  stage_tile(tileA, atom + (size_t)tgt * 1024, lane);
  zero_tile(tileZ, lane);
  int co = lane & 15, hi = lane >> 4;

  // z = elu(node_out[src] * conv3x3(atom[tgt], edge_w)), staged into tileZ
  v16h be[5];
#pragma unroll
  for (int s = 0; s < 5; ++s) be[s] = load_bfrag(fragbuf, 5 + s, lane);
  lds_fence();
#pragma unroll
  for (int mt = 0; mt < 4; ++mt) {
    v8f acc = zero8f();
#pragma unroll
    for (int s = 0; s < 5; ++s)
      acc = wmma_f16(load_a_frag(tileA, lane, mt, s), be[s], acc);
    int pos0 = mt * 16 + hi * 8;
    const float* tn = node_out + ((size_t)src * 16 + co) * 64 + pos0;
    float4 t0 = *(const float4*)tn;
    float4 t1 = *(const float4*)(tn + 4);
    float thn[8] = {t0.x, t0.y, t0.z, t0.w, t1.x, t1.y, t1.z, t1.w};
#pragma unroll
    for (int r = 0; r < 8; ++r) {
      float zv = elu_f(thn[r] * acc[r]);
      int pos = pos0 + r;
      int y = pos >> 3, x = pos & 7;
      tileZ[((y + 1) * 10 + (x + 1)) * 16 + co] = (_Float16)zv;
    }
  }

  // t = conv3x3(z, lin_w) + b; msg = sigmoid(t[:16]) * softplus(t[16:32])
  v16h b0[5], b1[5];
#pragma unroll
  for (int s = 0; s < 5; ++s) {
    b0[s] = load_bfrag(fragbuf, 10 + s, lane);
    b1[s] = load_bfrag(fragbuf, 15 + s, lane);
  }
  float bflt = lin_b[co];
  float bcor = lin_b[co + 16];
  lds_fence();
#pragma unroll
  for (int mt = 0; mt < 4; ++mt) {
    v8f a0 = zero8f(), a1 = zero8f();
#pragma unroll
    for (int s = 0; s < 5; ++s) {
      v16h a = load_a_frag(tileZ, lane, mt, s);
      a0 = wmma_f16(a, b0[s], a0);
      a1 = wmma_f16(a, b1[s], a1);
    }
    int pos0 = mt * 16 + hi * 8;
    float* dst = accum + ((size_t)src * 16 + co) * 64 + pos0;
#pragma unroll
    for (int r = 0; r < 8; ++r) {
      float msg = sigmoid_f(a0[r] + bflt) * softplus_f(a1[r] + bcor);
      unsafeAtomicAdd(dst + r, msg);   // global_atomic_add_f32
    }
  }
}

// ---------------- epilogue ----------------
__global__ void zero_kernel(float* __restrict__ p, int n4) {
  float4 z = make_float4(0.f, 0.f, 0.f, 0.f);
  for (int i = blockIdx.x * blockDim.x + threadIdx.x; i < n4;
       i += gridDim.x * blockDim.x)
    ((float4*)p)[i] = z;
}

__global__ void final_kernel(const float* __restrict__ atom,
                             const float* __restrict__ gamma,
                             const float* __restrict__ beta,
                             float* __restrict__ out, int total) {
  for (int i = blockIdx.x * blockDim.x + threadIdx.x; i < total;
       i += gridDim.x * blockDim.x) {
    int c = (i >> 6) & 15;
    float scale = gamma[c] * rsqrtf(1.f + 1e-5f);
    float o  = atom[i] + out[i];            // output = atom + scatter(msg)
    out[i] = softplus_f(atom[i] + (o * scale + beta[c]));
  }
}

// ---------------- launch ----------------
extern "C" void kernel_launch(void* const* d_in, const int* in_sizes, int n_in,
                              void* d_out, int out_size, void* d_ws, size_t ws_size,
                              hipStream_t stream) {
  const float* atom   = (const float*)d_in[0];
  const int*   esrc   = (const int*)d_in[1];
  const int*   etgt   = (const int*)d_in[2];
  const float* edge_w = (const float*)d_in[3];
  const float* node_w = (const float*)d_in[4];
  const float* lin_w  = (const float*)d_in[5];
  const float* lin_b  = (const float*)d_in[6];
  const float* gamma  = (const float*)d_in[7];
  const float* beta   = (const float*)d_in[8];
  float* out = (float*)d_out;

  int N = in_sizes[0] / 1024;   // (N,16,8,8)
  int E = in_sizes[1];

  float*    node_out = (float*)d_ws;                                   // N*1024 f32
  _Float16* fragbuf  = (_Float16*)((char*)d_ws +
                        (size_t)N * 1024 * sizeof(float));             // 20*32*16 f16

  frag_setup_kernel<<<1, 32, 0, stream>>>(edge_w, node_w, lin_w, fragbuf);
  zero_kernel<<<2048, 256, 0, stream>>>(out, out_size / 4);            // d_out = accum
  node_conv_kernel<<<(N + 7) / 8, 256, 0, stream>>>(atom, fragbuf, node_out, N);
  edge_kernel<<<(E + 7) / 8, 256, 0, stream>>>(atom, esrc, etgt, node_out,
                                               lin_b, fragbuf, out, E);
  final_kernel<<<4096, 256, 0, stream>>>(atom, gamma, beta, out, out_size);
}
